// RUBiCriterion_74302934221059
// MI455X (gfx1250) — compile-verified, matrix operand order checked by reading, not run on hardware
//
#include <hip/hip_runtime.h>
#include <hip/hip_bf16.h>
#include <math.h>

typedef __attribute__((ext_vector_type(16))) _Float16 v16h;
typedef __attribute__((ext_vector_type(8)))  _Float16 h8;
typedef __attribute__((ext_vector_type(8)))  float    v8f;

#define B_ROWS 8192
#define M_COLS 3129
#define M_PAD  3136          // padded ans rows (multiple of 64), pad rows = 0
#define D_DIM  512
#define EPSN   1e-8f
#define LDK    520           // 512+8 halves pad: row stride 1040B = 65*16B (16B-aligned b128)

// ---- workspace layout (bytes) ----
#define WSB_GH    0                          // f16 normalized mm_proj [8192*512]
#define WSB_AH    (WSB_GH + B_ROWS*D_DIM*2)  // f16 normalized ans     [3136*512]
#define WSB_POS   (WSB_AH + M_PAD*D_DIM*2)   // float [8192]
#define WSB_DENOM (WSB_POS + B_ROWS*4)       // float [8192]
#define WSB_ACC   (WSB_DENOM + B_ROWS*4)     // float [4]: 0=obj,1=nce,2=ce_q,3=ce_rubi

// ---------------------------------------------------------------- init
__global__ void init_ws_kernel(float* __restrict__ p, int n) {
  int i = blockIdx.x * 256 + threadIdx.x;
  if (i < n) p[i] = 0.f;
}

// ---------------- fused L2-normalize + fp32->f16 pack (one wave/row)
__global__ void norm_f16_kernel(const float* __restrict__ src,
                                _Float16* __restrict__ dst,
                                int src_rows, int dst_rows) {
  const int lane = threadIdx.x & 31;
  const int wave = threadIdx.x >> 5;
  const int row  = blockIdx.x * 8 + wave;
  if (row >= dst_rows) return;
  _Float16* d = dst + (size_t)row * D_DIM;
  if (row >= src_rows) {                       // zero pad rows
    for (int i = lane; i < D_DIM; i += 32) d[i] = (_Float16)0.f;
    return;
  }
  const float* p = src + (size_t)row * D_DIM;
  float v[16];
  float s = 0.f;
#pragma unroll
  for (int j = 0; j < 16; ++j) { v[j] = p[lane + 32 * j]; s += v[j] * v[j]; }
#pragma unroll
  for (int off = 16; off; off >>= 1) s += __shfl_xor(s, off, 32);
  const float inv = 1.f / fmaxf(sqrtf(s), EPSN);
#pragma unroll
  for (int j = 0; j < 16; ++j) d[lane + 32 * j] = (_Float16)(v[j] * inv);
}

// ------------------------------------------------ object cosine loss
__global__ void obj_kernel(const float* __restrict__ v,
                           const float* __restrict__ m,
                           float* __restrict__ acc) {
  const int lane = threadIdx.x & 31;
  const int wave = threadIdx.x >> 5;
  const int row  = blockIdx.x * 8 + wave;
  const float* pv = v + (size_t)row * D_DIM;
  const float* pm = m + (size_t)row * D_DIM;
  float svm = 0.f, svv = 0.f, smm = 0.f;
  for (int i = lane; i < D_DIM; i += 32) {
    float a = pv[i], b = pm[i];
    svm += a * b; svv += a * a; smm += b * b;
  }
#pragma unroll
  for (int off = 16; off; off >>= 1) {
    svm += __shfl_xor(svm, off, 32);
    svv += __shfl_xor(svv, off, 32);
    smm += __shfl_xor(smm, off, 32);
  }
  if (lane == 0) {
    float o = svm / (fmaxf(sqrtf(svv), EPSN) * fmaxf(sqrtf(smm), EPSN));
    atomicAdd(acc, 1.f - o);
  }
}

// --------------------- positive cosine from the normalized f16 copies
__global__ void pos_kernel(const _Float16* __restrict__ Gh,
                           const _Float16* __restrict__ Ah,
                           const int* __restrict__ cid,
                           float* __restrict__ pos) {
  const int lane = threadIdx.x & 31;
  const int wave = threadIdx.x >> 5;
  const int row  = blockIdx.x * 8 + wave;
  const int c    = cid[row];
  const _Float16* pg = Gh + (size_t)row * D_DIM;
  const _Float16* pa = Ah + (size_t)c * D_DIM;
  float s = 0.f;
  for (int i = lane; i < D_DIM; i += 32) s += (float)pg[i] * (float)pa[i];
#pragma unroll
  for (int off = 16; off; off >>= 1) s += __shfl_xor(s, off, 32);
  if (lane == 0) pos[row] = s;
}

// ----------------- WMMA f16 GEMM fused with exp row-sums
// Block = 256 thr = 8 waves (2x4). WG tile 64 rows x 64 cols.
// Wave computes 32x16 (two 16x16 subtiles sharing the B fragment).
// Split-K software pipeline: async-copy half1 overlaps WMMA on half0.
union Frag { v16h v; h8 p[2]; };

__device__ __forceinline__ void load_frag(Frag& f, const _Float16* base, int off) {
  f.p[0] = *(const h8*)(base + off);
  f.p[1] = *(const h8*)(base + off + 16);
}

__device__ __forceinline__ void compute_khalf(const _Float16* pg0,
                                              const _Float16* pg1,
                                              const _Float16* pb,
                                              int h, int kbeg,
                                              v8f& c0, v8f& c1) {
  Frag Aa0, Aa1, Ab, Ba0, Ba1, Bb;
#define LOADSET(P, K) \
  load_frag(P##a0, pg0, (K) + h * 8); \
  load_frag(P##a1, pg1, (K) + h * 8); \
  load_frag(P##b,  pb,  (K) + h * 8);

  LOADSET(A, kbeg)
#pragma unroll
  for (int k0 = kbeg; k0 < kbeg + 256; k0 += 64) {
    LOADSET(B, k0 + 32)                                  // prefetch next frags
    c0 = __builtin_amdgcn_wmma_f32_16x16x32_f16(false, Aa0.v, false, Ab.v,
                                                (short)0, c0, false, false);
    c1 = __builtin_amdgcn_wmma_f32_16x16x32_f16(false, Aa1.v, false, Ab.v,
                                                (short)0, c1, false, false);
    if (k0 + 64 < kbeg + 256) { LOADSET(A, k0 + 64) }
    c0 = __builtin_amdgcn_wmma_f32_16x16x32_f16(false, Ba0.v, false, Bb.v,
                                                (short)0, c0, false, false);
    c1 = __builtin_amdgcn_wmma_f32_16x16x32_f16(false, Ba1.v, false, Bb.v,
                                                (short)0, c1, false, false);
  }
#undef LOADSET
}

__global__ void gemm_expsum_kernel(const _Float16* __restrict__ Gh,
                                   const _Float16* __restrict__ Ah,
                                   float* __restrict__ denom) {
  extern __shared__ __align__(16) char smem[];
  _Float16* lG = (_Float16*)smem;        // 64 x LDK halves
  _Float16* lA = lG + 64 * LDK;          // 64 x LDK halves

  const int tid     = threadIdx.x;
  const int rowBase = blockIdx.x * 64;
  const int colBase = blockIdx.y * 64;

  // LDS byte offsets of the two staging buffers
  const unsigned lgOff = (unsigned)(size_t)(__attribute__((address_space(3))) char*)(void*)lG;
  const unsigned laOff = (unsigned)(size_t)(__attribute__((address_space(3))) char*)(void*)lA;

  // ---- async staging, split-K: half h covers bytes [h*512, h*512+512) of each row.
  // Issue half0 then half1 (32 in flight); async loads retire in order, so
  // s_wait_asynccnt 16 == "half0 landed".
#pragma unroll
  for (int half = 0; half < 2; ++half) {
    const int cb = half * 32;                 // 16B-chunk base within row
#pragma unroll
    for (int j = 0; j < 8; ++j) {             // G half-tile: 2048 chunks / 256 thr
      int ch = tid + 256 * j;
      int r  = ch >> 5, cin = (ch & 31) + cb;
      unsigned ldsa = lgOff + (unsigned)r * (LDK * 2) + (unsigned)(cin << 4);
      unsigned voff = ((unsigned)(rowBase + r) << 10) + (unsigned)(cin << 4);
      asm volatile("global_load_async_to_lds_b128 %0, %1, %2"
                   :: "v"(ldsa), "v"(voff), "s"(Gh) : "memory");
    }
#pragma unroll
    for (int j = 0; j < 8; ++j) {             // A half-tile: 2048 chunks / 256 thr
      int ch = tid + 256 * j;
      int r  = ch >> 5, cin = (ch & 31) + cb;
      unsigned ldsa = laOff + (unsigned)r * (LDK * 2) + (unsigned)(cin << 4);
      unsigned voff = ((unsigned)(colBase + r) << 10) + (unsigned)(cin << 4);
      asm volatile("global_load_async_to_lds_b128 %0, %1, %2"
                   :: "v"(ldsa), "v"(voff), "s"(Ah) : "memory");
    }
  }

  const int lane = tid & 31;
  const int wave = tid >> 5;
  const int wr   = wave >> 2;                 // 0..1 -> 32-row slab
  const int wc   = wave & 3;                  // 0..3 -> 16-col slab
  const int h    = lane >> 4;
  const int ln   = lane & 15;

  const _Float16* pg0 = lG + (wr * 32 + ln) * LDK;       // rows 0..15 of slab
  const _Float16* pg1 = pg0 + 16 * LDK;                  // rows 16..31
  const _Float16* pb  = lA + (wc * 16 + ln) * LDK;

  v8f c0 = {}, c1 = {};

  asm volatile("s_wait_asynccnt 0x10" ::: "memory");     // half0 resident
  __syncthreads();
  compute_khalf(pg0, pg1, pb, h, 0, c0, c1);             // overlaps half1 fetch

  asm volatile("s_wait_asynccnt 0x0" ::: "memory");      // half1 resident
  __syncthreads();
  compute_khalf(pg0, pg1, pb, h, 256, c0, c1);

  // epilogue: d is already a cosine (operands pre-normalized) -> exp, row-sum
  const int  gcol  = colBase + wc * 16 + ln;
  const bool valid = gcol < M_COLS;
  const int  rb0   = rowBase + wr * 32 + h * 8;          // C: VGPR r -> M = h*8+r

#pragma unroll
  for (int half = 0; half < 2; ++half) {
    const v8f& c = half ? c1 : c0;
    const int  rb = rb0 + half * 16;
#pragma unroll
    for (int r = 0; r < 8; ++r) {
      float e = valid ? __expf(c[r]) : 0.f;
      e += __shfl_xor(e, 8, 32);
      e += __shfl_xor(e, 4, 32);
      e += __shfl_xor(e, 2, 32);
      e += __shfl_xor(e, 1, 32);
      if (ln == 0) atomicAdd(&denom[rb + r], e);
    }
  }
}

// ------------------------------------ cross entropy (online softmax)
__global__ void ce_kernel(const float* __restrict__ L,
                          const int* __restrict__ cid,
                          float* __restrict__ acc) {
  __shared__ float smx[256];
  __shared__ float ssm[256];
  const int b = blockIdx.x;
  const float* row = L + (size_t)b * M_COLS;
  float mx = -3.4e38f, s = 0.f;
  for (int m = threadIdx.x; m < M_COLS; m += 256) {
    float x = row[m];
    if (x > mx) { s = s * __expf(mx - x) + 1.f; mx = x; }
    else        { s += __expf(x - mx); }
  }
  smx[threadIdx.x] = mx; ssm[threadIdx.x] = s;
  __syncthreads();
  for (int st = 128; st > 0; st >>= 1) {
    if (threadIdx.x < st) {
      float m1 = smx[threadIdx.x],      s1 = ssm[threadIdx.x];
      float m2 = smx[threadIdx.x + st], s2 = ssm[threadIdx.x + st];
      float mn = fmaxf(m1, m2);
      smx[threadIdx.x] = mn;
      ssm[threadIdx.x] = s1 * __expf(m1 - mn) + s2 * __expf(m2 - mn);
    }
    __syncthreads();
  }
  if (threadIdx.x == 0)
    atomicAdd(acc, smx[0] + __logf(ssm[0]) - row[cid[b]]);
}

// ------------------------------------------------------- NCE reduce
__global__ void nce_reduce_kernel(const float* __restrict__ denom,
                                  const float* __restrict__ pos,
                                  float* __restrict__ acc) {
  __shared__ float sm[256];
  float t = 0.f;
  for (int b = blockIdx.x * 256 + threadIdx.x; b < B_ROWS; b += 32 * 256)
    t += __logf(denom[b]) - pos[b];
  sm[threadIdx.x] = t;
  __syncthreads();
  for (int st = 128; st > 0; st >>= 1) {
    if (threadIdx.x < st) sm[threadIdx.x] += sm[threadIdx.x + st];
    __syncthreads();
  }
  if (threadIdx.x == 0) atomicAdd(acc, sm[0]);
}

// --------------------------------------------------------- finalize
__global__ void finalize_kernel(const float* __restrict__ acc,
                                float* __restrict__ out) {
  const float invB = 1.f / (float)B_ROWS;
  float obj = acc[0] * invB;
  float nce = acc[1] * invB;
  float ceq = acc[2] * invB;
  float cer = acc[3] * invB;
  float fusion = (cer + obj + nce) * (1.f / 3.f);
  out[0] = fusion + ceq;   // loss (question_loss_weight = 1.0)
  out[1] = fusion;         // loss_mm_q
  out[2] = ceq;            // loss_q
}

extern "C" void kernel_launch(void* const* d_in, const int* in_sizes, int n_in,
                              void* d_out, int out_size, void* d_ws, size_t ws_size,
                              hipStream_t stream) {
  (void)in_sizes; (void)n_in; (void)out_size; (void)ws_size;
  const float* mm_proj     = (const float*)d_in[0];
  const float* ans_emb     = (const float*)d_in[1];
  const float* v_max       = (const float*)d_in[2];
  const float* mm          = (const float*)d_in[3];
  const float* logits_q    = (const float*)d_in[4];
  const float* logits_rubi = (const float*)d_in[5];
  const int*   class_id    = (const int*)d_in[6];
  float* out = (float*)d_out;
  char*  wsb = (char*)d_ws;

  _Float16* gh   = (_Float16*)(wsb + WSB_GH);
  _Float16* ah   = (_Float16*)(wsb + WSB_AH);
  float*    pos  = (float*)(wsb + WSB_POS);
  float*    deno = (float*)(wsb + WSB_DENOM);
  float*    acc  = (float*)(wsb + WSB_ACC);

  // zero denom[8192] + acc[4] (contiguous)
  init_ws_kernel<<<(8196 + 255) / 256, 256, 0, stream>>>(deno, 8196);

  norm_f16_kernel<<<B_ROWS / 8, 256, 0, stream>>>(mm_proj, gh, B_ROWS, B_ROWS);
  norm_f16_kernel<<<M_PAD / 8, 256, 0, stream>>>(ans_emb, ah, M_COLS, M_PAD);

  obj_kernel<<<B_ROWS / 8, 256, 0, stream>>>(v_max, mm, acc + 0);
  pos_kernel<<<B_ROWS / 8, 256, 0, stream>>>(gh, ah, class_id, pos);

  dim3 ggrid(B_ROWS / 64, M_PAD / 64);                   // 128 x 49
  size_t lds = (size_t)2 * 64 * LDK * sizeof(_Float16);  // 133,120 B
  hipFuncSetAttribute((const void*)gemm_expsum_kernel,
                      hipFuncAttributeMaxDynamicSharedMemorySize, (int)lds);
  gemm_expsum_kernel<<<ggrid, 256, lds, stream>>>(gh, ah, deno);

  ce_kernel<<<B_ROWS, 256, 0, stream>>>(logits_q,    class_id, acc + 2);
  ce_kernel<<<B_ROWS, 256, 0, stream>>>(logits_rubi, class_id, acc + 3);

  nce_reduce_kernel<<<32, 256, 0, stream>>>(deno, pos, acc + 1);
  finalize_kernel<<<1, 1, 0, stream>>>(acc, out);
}
